// MultiHeadAttention_81518479278491
// MI455X (gfx1250) — matrix operand order repaired and checked by reading, not compiled, not run on hardware
//
#include <hip/hip_runtime.h>

// MultiHeadAttention forward for MI455X (gfx1250, wave32, WMMA).
// Pipeline: f32->bf16 convert (W pre-transposed), 3x WMMA GEMM (QKV proj,
// bias, 1/sqrt(d) folded into Q; V stored d-major), flash-attention with
// v_wmma_f32_16x16x32_bf16 + online softmax.
// Data movement: double-buffered LDS tiles filled with gfx1250 async
// global->LDS copies (ASYNCcnt); all WMMA fragment reads are contiguous
// ds_load_b128s because transposes are folded into the global layouts.
// Workspace requirement: ~33.5 MB.

typedef __attribute__((ext_vector_type(16))) __bf16 v16bf;
typedef __attribute__((ext_vector_type(8)))  float  v8f;
typedef __attribute__((ext_vector_type(4)))  int    v4i;
typedef __attribute__((address_space(1))) v4i gv4i;   // global int4
typedef __attribute__((address_space(3))) v4i lv4i;   // LDS int4

union ABFrag { v16bf v; unsigned int u[8]; };

#if defined(__has_builtin)
#if __has_builtin(__builtin_amdgcn_global_load_async_to_lds_b128)
#define HAVE_ASYNC_LDS 1
#endif
#endif

__device__ __forceinline__ void tile_copy16(const unsigned short* g,
                                            unsigned short* l) {
#ifdef HAVE_ASYNC_LDS
  __builtin_amdgcn_global_load_async_to_lds_b128((gv4i*)g, (lv4i*)l, 0, 0);
#else
  *(uint4*)l = *(const uint4*)g;
#endif
}

__device__ __forceinline__ void wait_async_all() {
#ifdef HAVE_ASYNC_LDS
#if __has_builtin(__builtin_amdgcn_s_wait_asynccnt)
  __builtin_amdgcn_s_wait_asynccnt(0);
#else
  asm volatile("s_wait_asynccnt 0" ::: "memory");
#endif
#endif
}

__device__ __forceinline__ unsigned short f2bf(float f) {
  unsigned int u = __float_as_uint(f);
  u += 0x7FFFu + ((u >> 16) & 1u);   // round to nearest even
  return (unsigned short)(u >> 16);
}

// ---------------------------------------------------------------- convert ---
__global__ __launch_bounds__(256) void cvt_f32_bf16_x4(
    const float4* __restrict__ in, ushort4* __restrict__ out, int n4) {
  int i = blockIdx.x * 256 + threadIdx.x;
  if (i < n4) {
    float4 f = in[i];
    ushort4 o;
    o.x = f2bf(f.x); o.y = f2bf(f.y); o.z = f2bf(f.z); o.w = f2bf(f.w);
    out[i] = o;
  }
}

// W[512][512] (k,n) -> bf16 Wt[512][512] (n,k), 32x32 LDS-tiled transpose.
__global__ __launch_bounds__(256) void cvt_transpose_bf16(
    const float* __restrict__ in, unsigned short* __restrict__ out) {
  __shared__ float t[32][33];
  const int lx = threadIdx.x & 31, ly = threadIdx.x >> 5;   // 32x8
  const int kb = blockIdx.y * 32, nb = blockIdx.x * 32;
#pragma unroll
  for (int i = 0; i < 32; i += 8)
    t[ly + i][lx] = in[(kb + ly + i) * 512 + nb + lx];
  __syncthreads();
#pragma unroll
  for (int i = 0; i < 32; i += 8)
    out[(nb + ly + i) * 512 + kb + lx] = f2bf(t[lx][ly + i]);
}

// ------------------------------------------------------------------- GEMM ---
// C[M=8192, N=512] = X[8192,512] @ W + bias, W given transposed [n][k].
// trans==0: Out bf16 [b*8+h][s][d]   (Q, K)
// trans==1: Out bf16 [b*8+h][d][s]   (V, d-major for the P.V WMMA B operand)
#define TM 128
#define TN 64
#define TK 32

__global__ __launch_bounds__(256) void qkv_gemm(
    const unsigned short* __restrict__ Xb,   // bf16 [8192][512]
    const unsigned short* __restrict__ Wt,   // bf16 [512][512]  (n,k)
    const float* __restrict__ bias,          // [512]
    unsigned short* __restrict__ Out,
    float scale, int trans) {
  __shared__ __align__(16) unsigned short As[2][TM][TK];  // [m][k]  16 KB
  __shared__ __align__(16) unsigned short Bs[2][TN][TK];  // [n][k]   8 KB

  const int tid  = threadIdx.x;
  const int lane = tid & 31;
  const int wave = tid >> 5;        // 0..7
  const int hl   = lane >> 4;       // K-half select (WMMA 16-bit layout)
  const int l16  = lane & 15;
  const int Mbase = blockIdx.x * TM;
  const int Nbase = blockIdx.y * TN;

  // Cooperative loader indices (16B per async op per thread).
  const int ar = tid >> 2, ac = (tid & 3) * 8;   // A rows ar, ar+64
  const int nr = tid >> 2, nc = (tid & 3) * 8;   // B [n][k]

  auto issue_tile = [&](int k0, int buf) {
    tile_copy16(&Xb[(Mbase + ar) * 512 + k0 + ac],      &As[buf][ar][ac]);
    tile_copy16(&Xb[(Mbase + 64 + ar) * 512 + k0 + ac], &As[buf][64 + ar][ac]);
    tile_copy16(&Wt[(Nbase + nr) * 512 + k0 + nc],      &Bs[buf][nr][nc]);
  };

  v8f acc[4];
#pragma unroll
  for (int nt = 0; nt < 4; ++nt)
#pragma unroll
    for (int r = 0; r < 8; ++r) acc[nt][r] = 0.0f;

  issue_tile(0, 0);

  for (int k0 = 0, it = 0; k0 < 512; k0 += TK, ++it) {
    const int buf = it & 1;
    wait_async_all();        // this wave's async writes to LDS landed
    __syncthreads();         // everyone's writes visible; prev reads done
    if (k0 + TK < 512) issue_tile(k0 + TK, buf ^ 1);  // prefetch next tile

    // A fragment: lane row m = l16; VGPR i holds K pair per 16-bit A layout.
    ABFrag a;
    const int m = wave * 16 + l16;
#pragma unroll
    for (int i = 0; i < 8; ++i) {
      int kk = ((i < 4) ? 2 * i : 2 * i + 8) + hl * 8;
      a.u[i] = *(const unsigned int*)&As[buf][m][kk];
    }
#pragma unroll
    for (int nt = 0; nt < 4; ++nt) {
      ABFrag b;
      const int n = nt * 16 + l16;
#pragma unroll
      for (int j = 0; j < 8; ++j) {
        int kk = ((j < 4) ? 2 * j : 2 * j + 8) + hl * 8;
        b.u[j] = *(const unsigned int*)&Bs[buf][n][kk];
      }
      acc[nt] = __builtin_amdgcn_wmma_f32_16x16x32_bf16(
          false, a.v, false, b.v, (short)0, acc[nt], false, false);
    }
  }

  // Epilogue: +bias, *scale, write bf16.
#pragma unroll
  for (int nt = 0; nt < 4; ++nt) {
    const int n  = Nbase + nt * 16 + l16;
    const float bn = bias[n];
    const int hh = n >> 6, dd = n & 63;
    const int mg0 = Mbase + wave * 16 + 8 * hl;   // first C row of this lane
    const int bb = mg0 >> 12;
    const int bh = (bb << 3) | hh;
    if (!trans) {
#pragma unroll
      for (int r = 0; r < 8; ++r) {
        int ss = (mg0 + r) & 4095;
        Out[((size_t)bh * 4096 + ss) * 64 + dd] = f2bf((acc[nt][r] + bn) * scale);
      }
    } else {
      // 8 consecutive s per lane -> single 16B store into [bh][d][s].
      unsigned int pk[4];
#pragma unroll
      for (int r2 = 0; r2 < 4; ++r2) {
        unsigned int lo = f2bf((acc[nt][2 * r2]     + bn) * scale);
        unsigned int hi = f2bf((acc[nt][2 * r2 + 1] + bn) * scale);
        pk[r2] = lo | (hi << 16);
      }
      int ss = mg0 & 4095;
      *(uint4*)&Out[((size_t)bh * 64 + dd) * 4096 + ss] = *(uint4*)pk;
    }
  }
}

// -------------------------------------------------------------- attention ---
// One block per (b*8+h, 128-query tile). 8 waves x 16 query rows.
__global__ __launch_bounds__(256) void attn_kernel(
    const unsigned short* __restrict__ Qb,  // bf16 [16][4096][64] (pre-scaled)
    const unsigned short* __restrict__ Kb,  // bf16 [16][4096][64]
    const unsigned short* __restrict__ Vt,  // bf16 [16][64][4096] (d-major)
    float* __restrict__ Out) {              // f32 [2][4096][512]
  __shared__ __align__(16) unsigned short Ks[2][32][64];   // 8 KB  [kv][d]
  __shared__ __align__(16) unsigned short Vst[2][64][32];  // 8 KB  [d][kv]
  __shared__ __align__(16) unsigned short Ps[8][16][32];   // 8 KB  per-wave P

  const int tid  = threadIdx.x;
  const int lane = tid & 31;
  const int wave = tid >> 5;
  const int hl   = lane >> 4;
  const int l16  = lane & 15;
  const int bh   = blockIdx.y;               // 0..15  (= b*8 + h)
  const int q0   = blockIdx.x * 128;

  const unsigned short* Qh = Qb + (size_t)bh * 4096 * 64;
  const unsigned short* Kh = Kb + (size_t)bh * 4096 * 64;
  const unsigned short* Vh = Vt + (size_t)bh * 64 * 4096;

  // Cooperative loader indices (4 KB tile = 256 threads x 16 B each).
  const int kr = tid >> 3, kc = (tid & 7) * 8;   // K [kv(32)][d(64)]
  const int vr = tid >> 2, vc = (tid & 3) * 8;   // V [d(64)][kv(32)]
  auto issue_tile = [&](int kv, int buf) {
    tile_copy16(&Kh[(kv + kr) * 64 + kc],        &Ks[buf][kr][kc]);
    tile_copy16(&Vh[(size_t)vr * 4096 + kv + vc], &Vst[buf][vr][vc]);
  };

  // Q fragments (two 32-wide d-chunks), loaded once. A-layout row = l16.
  const int qrow = q0 + wave * 16 + l16;
  ABFrag qf[2];
#pragma unroll
  for (int c = 0; c < 2; ++c)
#pragma unroll
    for (int i = 0; i < 8; ++i) {
      int kk = c * 32 + ((i < 4) ? 2 * i : 2 * i + 8) + hl * 8;
      qf[c].u[i] = *(const unsigned int*)&Qh[qrow * 64 + kk];
    }

  float mrow[8], lrow[8];
  v8f o[4];
#pragma unroll
  for (int r = 0; r < 8; ++r) { mrow[r] = -__builtin_inff(); lrow[r] = 0.0f; }
#pragma unroll
  for (int nt = 0; nt < 4; ++nt)
#pragma unroll
    for (int r = 0; r < 8; ++r) o[nt][r] = 0.0f;

  issue_tile(0, 0);

  for (int kv = 0, it = 0; kv < 4096; kv += 32, ++it) {
    const int buf = it & 1;
    wait_async_all();
    __syncthreads();                       // tile ready; prev buf reads done
    if (kv + 32 < 4096) issue_tile(kv + 32, buf ^ 1);   // overlap next fetch

    // Scores: S[16 q][32 kv] as two 16x16 WMMA tiles (K-dim = d = 64).
    v8f s0, s1;
#pragma unroll
    for (int r = 0; r < 8; ++r) { s0[r] = 0.0f; s1[r] = 0.0f; }
#pragma unroll
    for (int c = 0; c < 2; ++c) {
      ABFrag b0, b1;
#pragma unroll
      for (int j = 0; j < 8; ++j) {
        int kk = c * 32 + ((j < 4) ? 2 * j : 2 * j + 8) + hl * 8;           // d index pair
        b0.u[j] = *(const unsigned int*)&Ks[buf][l16][kk];       // kv 0..15
        b1.u[j] = *(const unsigned int*)&Ks[buf][16 + l16][kk];  // kv 16..31
      }
      s0 = __builtin_amdgcn_wmma_f32_16x16x32_bf16(false, qf[c].v, false, b0.v,
                                                   (short)0, s0, false, false);
      s1 = __builtin_amdgcn_wmma_f32_16x16x32_bf16(false, qf[c].v, false, b1.v,
                                                   (short)0, s1, false, false);
    }

    // Online softmax over the 32 new columns (row = r + 8*hl).
#pragma unroll
    for (int r = 0; r < 8; ++r) {
      float v = fmaxf(s0[r], s1[r]);
      v = fmaxf(v, __shfl_xor(v, 1, 32));
      v = fmaxf(v, __shfl_xor(v, 2, 32));
      v = fmaxf(v, __shfl_xor(v, 4, 32));
      v = fmaxf(v, __shfl_xor(v, 8, 32));
      float mnew = fmaxf(mrow[r], v);
      float corr = __expf(mrow[r] - mnew);
      float p0 = __expf(s0[r] - mnew);
      float p1 = __expf(s1[r] - mnew);
      float ps = p0 + p1;
      ps += __shfl_xor(ps, 1, 32);
      ps += __shfl_xor(ps, 2, 32);
      ps += __shfl_xor(ps, 4, 32);
      ps += __shfl_xor(ps, 8, 32);
      lrow[r] = lrow[r] * corr + ps;
      mrow[r] = mnew;
      Ps[wave][r + 8 * hl][l16]      = f2bf(p0);
      Ps[wave][r + 8 * hl][16 + l16] = f2bf(p1);
#pragma unroll
      for (int nt = 0; nt < 4; ++nt) o[nt][r] *= corr;
    }
    __syncthreads();  // make per-wave P slab visible across lanes

    // O += P[16x32] @ V[32x64] : A = P (LDS re-layout), B from Vst [d][kv].
    ABFrag pa;
#pragma unroll
    for (int i = 0; i < 8; ++i) {
      int kk = ((i < 4) ? 2 * i : 2 * i + 8) + hl * 8;   // kv pair
      pa.u[i] = *(const unsigned int*)&Ps[wave][l16][kk];
    }
#pragma unroll
    for (int nt = 0; nt < 4; ++nt) {
      ABFrag vb;
      const int n = nt * 16 + l16;                        // d column
#pragma unroll
      for (int j = 0; j < 8; ++j) {
        int kk = ((j < 4) ? 2 * j : 2 * j + 8) + hl * 8;                         // kv pair
        vb.u[j] = *(const unsigned int*)&Vst[buf][n][kk];
      }
      o[nt] = __builtin_amdgcn_wmma_f32_16x16x32_bf16(
          false, pa.v, false, vb.v, (short)0, o[nt], false, false);
    }
  }

  // Epilogue: normalize and store fp32 to [B, S, H].
  const int b = bh >> 3, h = bh & 7;
#pragma unroll
  for (int nt = 0; nt < 4; ++nt) {
    const int d = nt * 16 + l16;
#pragma unroll
    for (int r = 0; r < 8; ++r) {
      int qr = q0 + wave * 16 + r + 8 * hl;
      Out[((size_t)b * 4096 + qr) * 512 + h * 64 + d] = o[nt][r] / lrow[r];
    }
  }
}

// ----------------------------------------------------------------- launch ---
extern "C" void kernel_launch(void* const* d_in, const int* in_sizes, int n_in,
                              void* d_out, int out_size, void* d_ws, size_t ws_size,
                              hipStream_t stream) {
  const float* x  = (const float*)d_in[0];
  const float* Wq = (const float*)d_in[1];
  const float* bq = (const float*)d_in[2];
  const float* Wk = (const float*)d_in[3];
  const float* bk = (const float*)d_in[4];
  const float* Wv = (const float*)d_in[5];
  const float* bv = (const float*)d_in[6];
  float* out = (float*)d_out;

  const int M = 8192, H = 512;            // B*S, hidden
  unsigned short* xb  = (unsigned short*)d_ws;             // 8 MB
  unsigned short* wqb = xb  + (size_t)M * H;               // 512 KB each
  unsigned short* wkb = wqb + (size_t)H * H;
  unsigned short* wvb = wkb + (size_t)H * H;
  unsigned short* Qb  = wvb + (size_t)H * H;               // 8 MB each
  unsigned short* Kb  = Qb  + (size_t)16 * 4096 * 64;
  unsigned short* Vb  = Kb  + (size_t)16 * 4096 * 64;

  // fp32 -> bf16 conversion (x straight; W transposed to [n][k]).
  int nx4 = M * H / 4;
  cvt_f32_bf16_x4<<<nx4 / 256, 256, 0, stream>>>((const float4*)x, (ushort4*)xb, nx4);
  dim3 gt(H / 32, H / 32);
  cvt_transpose_bf16<<<gt, 256, 0, stream>>>(Wq, wqb);
  cvt_transpose_bf16<<<gt, 256, 0, stream>>>(Wk, wkb);
  cvt_transpose_bf16<<<gt, 256, 0, stream>>>(Wv, wvb);

  // QKV projections (1/sqrt(64) folded into Q; V stored d-major).
  dim3 gg(M / TM, H / TN);
  qkv_gemm<<<gg, 256, 0, stream>>>(xb, wqb, bq, Qb, 0.125f, 0);
  qkv_gemm<<<gg, 256, 0, stream>>>(xb, wkb, bk, Kb, 1.0f, 0);
  qkv_gemm<<<gg, 256, 0, stream>>>(xb, wvb, bv, Vb, 1.0f, 1);

  // Flash attention
  dim3 ga(4096 / 128, 16);
  attn_kernel<<<ga, 256, 0, stream>>>(Qb, Kb, Vb, out);
}
